// HybridHeadAttention_20255065768198
// MI455X (gfx1250) — matrix operand order repaired and checked by reading, not compiled, not run on hardware
//
#include <hip/hip_runtime.h>
#include <hip/hip_bf16.h>

typedef __attribute__((ext_vector_type(16))) __bf16 bf16x16;
typedef __attribute__((ext_vector_type(8)))  float  v8f;

// Pointer types matching the async-LDS builtin's parameters exactly:
// gcc-style int4 vector, with address space attached via typedef.
typedef int gv4i __attribute__((vector_size(16)));
typedef gv4i __attribute__((address_space(1))) gv4i_as1;   // global
typedef gv4i __attribute__((address_space(3))) gv4i_as3;   // LDS

// ---------------------------------------------------------------- feature probe
#if defined(__has_builtin)
#if __has_builtin(__builtin_amdgcn_global_load_async_to_lds_b128)
#define HAVE_ASYNC_COPY 1
#endif
#endif
#ifndef HAVE_ASYNC_COPY
#define HAVE_ASYNC_COPY 0
#endif

// ---------------------------------------------------------------- WMMA helpers

__device__ __forceinline__ v8f wmma_bf16(bf16x16 a, bf16x16 b, v8f c) {
  return __builtin_amdgcn_wmma_f32_16x16x32_bf16(false, a, false, b,
                                                 (short)0, c, false, false);
}

// A fragment 16x32 (M x K), src row-major [row][k], contiguous k order (not the ISA table order):
// lanes 0-15 : row=lane, K = 0..15;  lanes 16-31: row=lane-16, K = 16..31
__device__ __forceinline__ bf16x16 ldfrag_a(const __bf16* src, int ld) {
  const int lane = threadIdx.x & 31;
  const int r    = lane & 15;
  const int kb   = (lane >> 4) << 4;
  const __bf16* p = src + (size_t)r * ld + kb;
  bf16x16 f;
#pragma unroll
  for (int j = 0; j < 16; ++j) f[j] = p[j];
  return f;
}

// B fragment 32x16 (K x N), operand stored [N][K] (K contiguous per output col).
__device__ __forceinline__ bf16x16 ldfrag_b_nk(const __bf16* src, int ld) {
  const int lane = threadIdx.x & 31;
  const int n    = lane & 15;
  const int k0   = (lane >> 4) << 4;
  const __bf16* p = src + (size_t)n * ld + k0;
  bf16x16 f;
#pragma unroll
  for (int j = 0; j < 16; ++j) f[j] = p[j];
  return f;
}

// B fragment 32x16, operand stored [K][N] (N contiguous), e.g. V tile in LDS.
__device__ __forceinline__ bf16x16 ldfrag_b_kn(const __bf16* src, int ld) {
  const int lane = threadIdx.x & 31;
  const int n    = lane & 15;
  const int k0   = (lane >> 4) << 4;
  bf16x16 f;
#pragma unroll
  for (int j = 0; j < 16; ++j) f[j] = src[(size_t)(k0 + j) * ld + n];
  return f;
}

// ---------------------------------------------------------------- async LDS copy
// 16-byte global -> LDS copy. On CDNA5 uses global_load_async_to_lds_b128
// (ASYNCcnt path, no VGPR round-trip); falls back to a regular copy otherwise.
__device__ __forceinline__ void cp_lds_b128(void* lds_dst, const void* gsrc) {
#if HAVE_ASYNC_COPY
  gv4i_as1* g = (gv4i_as1*)(unsigned long long)gsrc;
  gv4i_as3* l = (gv4i_as3*)(unsigned int)(unsigned long long)lds_dst;
  __builtin_amdgcn_global_load_async_to_lds_b128(g, l, 0, 0);
#else
  *reinterpret_cast<uint4*>(lds_dst) = *reinterpret_cast<const uint4*>(gsrc);
#endif
}

// wave waits for its own async copies, then workgroup barrier
__device__ __forceinline__ void async_join_barrier() {
#if HAVE_ASYNC_COPY
#if __has_builtin(__builtin_amdgcn_s_wait_asynccnt)
  __builtin_amdgcn_s_wait_asynccnt(0);
#else
  asm volatile("s_wait_asynccnt 0x0" ::: "memory");
#endif
#endif
  __syncthreads();
}

// ---------------------------------------------------------------- constants
constexpr int B_  = 16;
constexpr int N_  = 1024;
constexpr int C_  = 768;
constexpr int H_  = 12;
constexpr int D_  = 64;
constexpr int NNYS = 6;
constexpr float SCALE_ = 0.125f;   // 64^-0.5

// ================================================================ 0) fp32 -> bf16
__global__ void cvt_f32_bf16_kernel(const float* __restrict__ src,
                                    __bf16* __restrict__ dst, int n4) {
  const int i = blockIdx.x * blockDim.x + threadIdx.x;
  if (i < n4) {
    const float4 f = reinterpret_cast<const float4*>(src)[i];
    __bf16* o = dst + (size_t)i * 4;
    o[0] = (__bf16)f.x; o[1] = (__bf16)f.y;
    o[2] = (__bf16)f.z; o[3] = (__bf16)f.w;
  }
}

// ================================================================ 1) QKV GEMM
// A = xb[16384,768] bf16, Bm = qkv_w bf16 [2304,768]; 128x128 block tile,
// 8 waves of 32x64, async-LDS staged. Writes scaled Q, K, V bf16 [B][H][N][D].
__global__ __launch_bounds__(256) void qkv_gemm_kernel(
    const __bf16* __restrict__ A, const __bf16* __restrict__ Bm,
    const float* __restrict__ bias,
    __bf16* __restrict__ Q, __bf16* __restrict__ K, __bf16* __restrict__ V) {
  constexpr int KD = 768;
  __shared__ __bf16 As[128][32];
  __shared__ __bf16 Bs[128][32];
  const int t    = threadIdx.x;
  const int m0   = blockIdx.x * 128;
  const int n0   = blockIdx.y * 128;
  const int wid  = t >> 5, lane = t & 31;
  const int wm   = (wid & 3) * 32;   // 4 waves along M
  const int wn   = (wid >> 2) * 64;  // 2 waves along N
  v8f acc[2][4] = {};
  const int lr = t >> 1, lk = (t & 1) * 16;   // 128 rows x 32 k, 16 elems/thread

  for (int k0 = 0; k0 < KD; k0 += 32) {
    const __bf16* ga = A  + (size_t)(m0 + lr) * KD + k0 + lk;
    const __bf16* gb = Bm + (size_t)(n0 + lr) * KD + k0 + lk;
    cp_lds_b128(&As[lr][lk],     ga);
    cp_lds_b128(&As[lr][lk + 8], ga + 8);
    cp_lds_b128(&Bs[lr][lk],     gb);
    cp_lds_b128(&Bs[lr][lk + 8], gb + 8);
    async_join_barrier();
    const bf16x16 a0 = ldfrag_a(&As[wm][0], 32);
    const bf16x16 a1 = ldfrag_a(&As[wm + 16][0], 32);
#pragma unroll
    for (int ni = 0; ni < 4; ++ni) {
      const bf16x16 bfr = ldfrag_b_nk(&Bs[wn + ni * 16][0], 32);
      acc[0][ni] = wmma_bf16(a0, bfr, acc[0][ni]);
      acc[1][ni] = wmma_bf16(a1, bfr, acc[1][ni]);
    }
    __syncthreads();
  }
  const int half = lane >> 4, lc = lane & 15;
#pragma unroll
  for (int mi = 0; mi < 2; ++mi)
#pragma unroll
    for (int ni = 0; ni < 4; ++ni) {
      const int rowb = m0 + wm + mi * 16;
      const int colb = n0 + wn + ni * 16;
#pragma unroll
      for (int r = 0; r < 8; ++r) {
        const int grow = rowb + r + half * 8;
        const int gcol = colb + lc;
        float val = acc[mi][ni][r] + bias[gcol];
        const int which = gcol / 768, cidx = gcol % 768;
        const int h = cidx >> 6, d = cidx & 63;
        if (which == 0) val *= SCALE_;
        __bf16* dst = (which == 0) ? Q : (which == 1) ? K : V;
        const int b = grow >> 10, n = grow & 1023;
        dst[((((size_t)b * H_ + h) << 10) + n) * D_ + d] = (__bf16)val;
      }
    }
}

// ================================================================ 2) landmark pooling
__global__ void pool_kernel(const __bf16* __restrict__ Q,
                            const __bf16* __restrict__ K,
                            float* __restrict__ qm, float* __restrict__ km) {
  const int d = threadIdx.x;                 // 64
  const int m = blockIdx.x, h = blockIdx.y, b = blockIdx.z;
  const size_t base = (((size_t)b * H_ + h) << 10) * D_;
  float sq = 0.f, sk = 0.f;
  for (int j = 0; j < 32; ++j) {
    const size_t off = base + (size_t)(m * 32 + j) * D_ + d;
    sq += (float)Q[off];
    sk += (float)K[off];
  }
  const size_t o = (((size_t)b * NNYS + h) * 32 + m) * D_ + d;
  qm[o] = sq * (1.f / 32.f);
  km[o] = sk * (1.f / 32.f);
}

// ================================================================ 3) Nystrom stage A
__global__ __launch_bounds__(256) void nys_stage_a_kernel(
    const __bf16* __restrict__ K, const __bf16* __restrict__ V,
    const float* __restrict__ qm, const float* __restrict__ km,
    float* __restrict__ Sc, float* __restrict__ T2) {
  __shared__ float qs[32][64];
  __shared__ float ks[32][64];
  __shared__ float cmax[32];
  __shared__ float xm[32][32];
  __shared__ float zm[32][32];
  __shared__ float s0[32][32];
  __shared__ float s1[32][32];
  __shared__ float s2[32][32];
  __shared__ float T1[32][66];
  __shared__ float rs[32], cs[32];
  __shared__ float norm_s;
  const int t = threadIdx.x;
  const int h = blockIdx.x, b = blockIdx.y;
  const size_t kv_base = (((size_t)b * H_ + h) << 10) * D_;
  const size_t lm_base = (((size_t)b * NNYS + h) * 32) * D_;
  float* ScRow = Sc + ((size_t)b * NNYS + h) * 32 * N_;

  for (int i = t; i < 2048; i += 256) {
    qs[i >> 6][i & 63] = qm[lm_base + i];
    ks[i >> 6][i & 63] = km[lm_base + i];
  }
  __syncthreads();
  // S_C = qm @ kn^T  (32 x 1024) -> global scratch
  for (int n = t; n < N_; n += 256) {
    float kr[64];
    const __bf16* kp = K + kv_base + (size_t)n * D_;
#pragma unroll
    for (int d2 = 0; d2 < 64; ++d2) kr[d2] = (float)kp[d2];
    for (int m = 0; m < 32; ++m) {
      float a = 0.f;
#pragma unroll
      for (int d2 = 0; d2 < 64; ++d2) a += qs[m][d2] * kr[d2];
      ScRow[(size_t)m * N_ + n] = a;
    }
  }
  __threadfence();
  __syncthreads();
  if (t < 32) {
    float mx = -1e30f;
    const float* row = ScRow + (size_t)t * N_;
    for (int n = 0; n < N_; ++n) mx = fmaxf(mx, row[n]);
    cmax[t] = mx;
  }
  __syncthreads();
  // T1 = expC @ v_aug  (32 x 65)
  {
    const int m = t >> 3, c = t & 7;
    const float cm = cmax[m];
    const float* row = ScRow + (size_t)m * N_;
    float a[8] = {};
    float esum = 0.f;
    for (int n = 0; n < N_; ++n) {
      const float e = __expf(row[n] - cm);
      const __bf16* vp = V + kv_base + (size_t)n * D_ + c * 8;
#pragma unroll
      for (int j = 0; j < 8; ++j) a[j] += e * (float)vp[j];
      if (c == 0) esum += e;
    }
#pragma unroll
    for (int j = 0; j < 8; ++j) T1[m][c * 8 + j] = a[j];
    if (c == 0) T1[m][64] = esum;
  }
  // expB = exp(max(qm@km^T - cmax, -88))
  for (int idx = t; idx < 1024; idx += 256) {
    const int m = idx >> 5, e2 = idx & 31;
    float a = 0.f;
#pragma unroll
    for (int d2 = 0; d2 < 64; ++d2) a += qs[m][d2] * ks[e2][d2];
    xm[m][e2] = __expf(fmaxf(a - cmax[m], -88.f));
  }
  __syncthreads();
  // pinv init: z = x^T / (max-rowsum * max-colsum)
  if (t < 32) {
    float s = 0.f;
    for (int e2 = 0; e2 < 32; ++e2) s += xm[t][e2];
    rs[t] = s;
  } else if (t < 64) {
    const int e2 = t - 32;
    float s = 0.f;
    for (int m = 0; m < 32; ++m) s += xm[m][e2];
    cs[e2] = s;
  }
  __syncthreads();
  if (t == 0) {
    float colv = rs[0], rowv = cs[0];
    for (int i = 1; i < 32; ++i) {
      colv = fmaxf(colv, rs[i]);
      rowv = fmaxf(rowv, cs[i]);
    }
    norm_s = 1.f / (colv * rowv);
  }
  __syncthreads();
  for (int idx = t; idx < 1024; idx += 256) {
    const int m = idx >> 5, e2 = idx & 31;
    zm[m][e2] = xm[e2][m] * norm_s;
  }
  __syncthreads();
  // 6 Newton-Schulz iterations: z = 0.25 z (13I - xz(15I - xz(7I - xz)))
  for (int it = 0; it < 6; ++it) {
    for (int idx = t; idx < 1024; idx += 256) {
      const int m = idx >> 5, e2 = idx & 31;
      float a = 0.f;
#pragma unroll
      for (int k2 = 0; k2 < 32; ++k2) a += xm[m][k2] * zm[k2][e2];
      s0[m][e2] = a;
    }
    __syncthreads();
    for (int idx = t; idx < 1024; idx += 256) {
      const int m = idx >> 5, e2 = idx & 31;
      float a = 0.f;
#pragma unroll
      for (int k2 = 0; k2 < 32; ++k2)
        a += s0[m][k2] * (((k2 == e2) ? 7.f : 0.f) - s0[k2][e2]);
      s1[m][e2] = ((m == e2) ? 15.f : 0.f) - a;
    }
    __syncthreads();
    for (int idx = t; idx < 1024; idx += 256) {
      const int m = idx >> 5, e2 = idx & 31;
      float a = 0.f;
#pragma unroll
      for (int k2 = 0; k2 < 32; ++k2) a += s0[m][k2] * s1[k2][e2];
      s2[m][e2] = ((m == e2) ? 13.f : 0.f) - a;
    }
    __syncthreads();
    for (int idx = t; idx < 1024; idx += 256) {
      const int m = idx >> 5, e2 = idx & 31;
      float a = 0.f;
#pragma unroll
      for (int k2 = 0; k2 < 32; ++k2) a += zm[m][k2] * s2[k2][e2];
      s1[m][e2] = 0.25f * a;
    }
    __syncthreads();
    for (int idx = t; idx < 1024; idx += 256) {
      const int m = idx >> 5, e2 = idx & 31;
      zm[m][e2] = s1[m][e2];
    }
    __syncthreads();
  }
  // T2 = z @ T1  (32 x 65)
  float* T2p = T2 + ((size_t)b * NNYS + h) * 32 * 65;
  for (int idx = t; idx < 32 * 65; idx += 256) {
    const int m = idx / 65, d2 = idx % 65;
    float a = 0.f;
#pragma unroll
    for (int k2 = 0; k2 < 32; ++k2) a += zm[m][k2] * T1[k2][d2];
    T2p[idx] = a;
  }
}

// ================================================================ 4) Nystrom stage B
__global__ __launch_bounds__(256) void nys_stage_b_kernel(
    const __bf16* __restrict__ Q, const float* __restrict__ km,
    const float* __restrict__ T2, __bf16* __restrict__ ctx) {
  __shared__ float kms[32][64];
  __shared__ float T2s[32][66];
  __shared__ float qsh[64][64];
  __shared__ float sc[64][33];
  __shared__ float den[64];
  const int t = threadIdx.x;
  const int chunk = blockIdx.x, h = blockIdx.y, b = blockIdx.z;
  const size_t q_base = ((((size_t)b * H_ + h) << 10) + chunk * 64) * D_;
  const size_t lm_base = (((size_t)b * NNYS + h) * 32) * D_;
  const float* T2p = T2 + ((size_t)b * NNYS + h) * 32 * 65;

  for (int i = t; i < 2048; i += 256) kms[i >> 6][i & 63] = km[lm_base + i];
  for (int i = t; i < 32 * 65; i += 256) T2s[i / 65][i % 65] = T2p[i];
  for (int i = t; i < 4096; i += 256) qsh[i >> 6][i & 63] = (float)Q[q_base + i];
  __syncthreads();
  const int r = t >> 2, p = t & 3;
  for (int m = p * 8; m < p * 8 + 8; ++m) {
    float a = 0.f;
#pragma unroll
    for (int d2 = 0; d2 < 64; ++d2) a += qsh[r][d2] * kms[m][d2];
    sc[r][m] = a;
  }
  __syncthreads();
  float mx = sc[r][0];
  for (int m = 1; m < 32; ++m) mx = fmaxf(mx, sc[r][m]);
  float num[16] = {};
  float dn = 0.f;
  for (int m = 0; m < 32; ++m) {
    const float e = __expf(sc[r][m] - mx);
#pragma unroll
    for (int j = 0; j < 16; ++j) num[j] += e * T2s[m][p * 16 + j];
    if (p == 0) dn += e * T2s[m][64];
  }
  if (p == 0) den[r] = fmaxf(dn, 1e-8f);
  __syncthreads();
  const float inv = 1.f / den[r];
  const int row = chunk * 64 + r;
  __bf16* outp = ctx + ((size_t)b * N_ + row) * C_ + h * D_ + p * 16;
#pragma unroll
  for (int j = 0; j < 16; ++j) outp[j] = (__bf16)(num[j] * inv);
}

// ================================================================ 5) flash attention (heads 6..11)
__global__ __launch_bounds__(256) void flash_attn_kernel(
    const __bf16* __restrict__ Q, const __bf16* __restrict__ K,
    const __bf16* __restrict__ V, __bf16* __restrict__ ctx) {
  __shared__ __bf16 Kb[64][64];
  __shared__ __bf16 Vb[64][64];
  __shared__ __bf16 Pl[8][16][64];
  const int t = threadIdx.x, lane = t & 31, wid = t >> 5;
  const int rb = blockIdx.x, h = NNYS + blockIdx.y, b = blockIdx.z;
  const size_t head_off = (((size_t)b * H_ + h) << 10) * D_;
  const int n0 = rb * 128 + wid * 16;

  const bf16x16 qa0 = ldfrag_a(Q + head_off + (size_t)n0 * D_, 64);
  const bf16x16 qa1 = ldfrag_a(Q + head_off + (size_t)n0 * D_ + 32, 64);

  float mrow[8], lrow[8];
#pragma unroll
  for (int r = 0; r < 8; ++r) { mrow[r] = -1e30f; lrow[r] = 0.f; }
  v8f acco[4] = {};
  const int cr = t >> 2, ck = (t & 3) * 16;
  const int half = lane >> 4, lc = lane & 15;

  for (int kb0 = 0; kb0 < N_; kb0 += 64) {
    const __bf16* kg = K + head_off + (size_t)(kb0 + cr) * D_ + ck;
    const __bf16* vg = V + head_off + (size_t)(kb0 + cr) * D_ + ck;
    cp_lds_b128(&Kb[cr][ck],     kg);
    cp_lds_b128(&Kb[cr][ck + 8], kg + 8);
    cp_lds_b128(&Vb[cr][ck],     vg);
    cp_lds_b128(&Vb[cr][ck + 8], vg + 8);
    async_join_barrier();
    // S = Q @ K^T for this 16x64 strip
    v8f s[4];
#pragma unroll
    for (int nt = 0; nt < 4; ++nt) {
      v8f z = {};
      z = wmma_bf16(qa0, ldfrag_b_nk(&Kb[nt * 16][0], 64), z);
      z = wmma_bf16(qa1, ldfrag_b_nk(&Kb[nt * 16][32], 64), z);
      s[nt] = z;
    }
    // online softmax update (rows live in lanes of same 16-group)
#pragma unroll
    for (int r = 0; r < 8; ++r) {
      float mx = fmaxf(fmaxf(s[0][r], s[1][r]), fmaxf(s[2][r], s[3][r]));
#pragma unroll
      for (int off = 8; off; off >>= 1) mx = fmaxf(mx, __shfl_xor(mx, off, 32));
      const float mnew = fmaxf(mrow[r], mx);
      const float alpha = __expf(mrow[r] - mnew);
      float rsum = 0.f;
#pragma unroll
      for (int nt = 0; nt < 4; ++nt) {
        const float e = __expf(s[nt][r] - mnew);
        s[nt][r] = e;
        rsum += e;
      }
#pragma unroll
      for (int off = 8; off; off >>= 1) rsum += __shfl_xor(rsum, off, 32);
      lrow[r] = lrow[r] * alpha + rsum;
      mrow[r] = mnew;
#pragma unroll
      for (int dt = 0; dt < 4; ++dt) acco[dt][r] *= alpha;
    }
    // C-layout P -> LDS (per-wave private region) so it can be re-read in A-layout
#pragma unroll
    for (int nt = 0; nt < 4; ++nt)
#pragma unroll
      for (int r = 0; r < 8; ++r)
        Pl[wid][r + half * 8][nt * 16 + lc] = (__bf16)s[nt][r];
    // acc += P @ V
#pragma unroll
    for (int dt = 0; dt < 4; ++dt)
#pragma unroll
      for (int ks = 0; ks < 2; ++ks) {
        const bf16x16 af = ldfrag_a(&Pl[wid][0][ks * 32], 64);
        const bf16x16 bfv = ldfrag_b_kn(&Vb[ks * 32][dt * 16], 64);
        acco[dt] = wmma_bf16(af, bfv, acco[dt]);
      }
    __syncthreads();
  }
#pragma unroll
  for (int dt = 0; dt < 4; ++dt)
#pragma unroll
    for (int r = 0; r < 8; ++r) {
      const int row = n0 + r + half * 8;
      const int d = dt * 16 + lc;
      const float val = acco[dt][r] / lrow[r];
      ctx[((size_t)b * N_ + row) * C_ + h * D_ + d] = (__bf16)val;
    }
}

// ================================================================ 6) output projection
__global__ __launch_bounds__(256) void proj_gemm_kernel(
    const __bf16* __restrict__ A, const __bf16* __restrict__ Bm,
    const float* __restrict__ bias, float* __restrict__ out) {
  constexpr int KD = 768;
  __shared__ __bf16 As[128][32];
  __shared__ __bf16 Bs[128][32];
  const int t   = threadIdx.x;
  const int m0  = blockIdx.x * 128;
  const int n0  = blockIdx.y * 128;
  const int wid = t >> 5, lane = t & 31;
  const int wm  = (wid & 3) * 32;
  const int wn  = (wid >> 2) * 64;
  v8f acc[2][4] = {};
  const int lr = t >> 1, lk = (t & 1) * 16;

  for (int k0 = 0; k0 < KD; k0 += 32) {
    const __bf16* ga = A  + (size_t)(m0 + lr) * KD + k0 + lk;
    const __bf16* gb = Bm + (size_t)(n0 + lr) * KD + k0 + lk;
    cp_lds_b128(&As[lr][lk],     ga);
    cp_lds_b128(&As[lr][lk + 8], ga + 8);
    cp_lds_b128(&Bs[lr][lk],     gb);
    cp_lds_b128(&Bs[lr][lk + 8], gb + 8);
    async_join_barrier();
    const bf16x16 a0 = ldfrag_a(&As[wm][0], 32);
    const bf16x16 a1 = ldfrag_a(&As[wm + 16][0], 32);
#pragma unroll
    for (int ni = 0; ni < 4; ++ni) {
      const bf16x16 bfr = ldfrag_b_nk(&Bs[wn + ni * 16][0], 32);
      acc[0][ni] = wmma_bf16(a0, bfr, acc[0][ni]);
      acc[1][ni] = wmma_bf16(a1, bfr, acc[1][ni]);
    }
    __syncthreads();
  }
  const int half = lane >> 4, lc = lane & 15;
#pragma unroll
  for (int mi = 0; mi < 2; ++mi)
#pragma unroll
    for (int ni = 0; ni < 4; ++ni) {
      const int rowb = m0 + wm + mi * 16;
      const int colb = n0 + wn + ni * 16;
#pragma unroll
      for (int r = 0; r < 8; ++r) {
        const int grow = rowb + r + half * 8;
        const int gcol = colb + lc;
        out[(size_t)grow * KD + gcol] = acc[mi][ni][r] + bias[gcol];
      }
    }
}

// ================================================================ launcher
extern "C" void kernel_launch(void* const* d_in, const int* in_sizes, int n_in,
                              void* d_out, int out_size, void* d_ws, size_t ws_size,
                              hipStream_t stream) {
  (void)in_sizes; (void)n_in; (void)out_size; (void)ws_size;
  const float* x      = (const float*)d_in[0];
  const float* qkv_w  = (const float*)d_in[1];
  const float* qkv_b  = (const float*)d_in[2];
  const float* proj_w = (const float*)d_in[3];
  const float* proj_b = (const float*)d_in[4];
  float* out = (float*)d_out;

  char* ws = (char*)d_ws;
  size_t off = 0;
  auto alloc = [&](size_t bytes) {
    void* p = ws + off;
    off = (off + bytes + 255) & ~(size_t)255;
    return p;
  };
  const size_t qkv_elems = (size_t)B_ * H_ * N_ * D_;        // 12.6M
  const size_t x_elems   = (size_t)B_ * N_ * C_;             // 12.6M
  const size_t wq_elems  = (size_t)3 * C_ * C_;              // 1.77M
  const size_t wp_elems  = (size_t)C_ * C_;                  // 0.59M
  __bf16* Q      = (__bf16*)alloc(qkv_elems * 2);
  __bf16* Kv     = (__bf16*)alloc(qkv_elems * 2);
  __bf16* Vv     = (__bf16*)alloc(qkv_elems * 2);
  __bf16* ctx    = (__bf16*)alloc(x_elems * 2);
  __bf16* xb     = (__bf16*)alloc(x_elems * 2);
  __bf16* wqkvb  = (__bf16*)alloc(wq_elems * 2);
  __bf16* wprojb = (__bf16*)alloc(wp_elems * 2);
  float*  qm     = (float*)alloc((size_t)B_ * NNYS * 32 * D_ * 4);
  float*  km     = (float*)alloc((size_t)B_ * NNYS * 32 * D_ * 4);
  float*  Sc     = (float*)alloc((size_t)B_ * NNYS * 32 * N_ * 4);
  float*  T2     = (float*)alloc((size_t)B_ * NNYS * 32 * 65 * 4);

  const int nx4 = (int)(x_elems / 4), nq4 = (int)(wq_elems / 4), np4 = (int)(wp_elems / 4);
  cvt_f32_bf16_kernel<<<(nx4 + 255) / 256, 256, 0, stream>>>(x, xb, nx4);
  cvt_f32_bf16_kernel<<<(nq4 + 255) / 256, 256, 0, stream>>>(qkv_w, wqkvb, nq4);
  cvt_f32_bf16_kernel<<<(np4 + 255) / 256, 256, 0, stream>>>(proj_w, wprojb, np4);

  qkv_gemm_kernel<<<dim3(128, 18), 256, 0, stream>>>(xb, wqkvb, qkv_b, Q, Kv, Vv);
  pool_kernel<<<dim3(32, NNYS, B_), 64, 0, stream>>>(Q, Kv, qm, km);
  nys_stage_a_kernel<<<dim3(NNYS, B_), 256, 0, stream>>>(Kv, Vv, qm, km, Sc, T2);
  nys_stage_b_kernel<<<dim3(16, NNYS, B_), 256, 0, stream>>>(Q, km, T2, ctx);
  flash_attn_kernel<<<dim3(8, NNYS, B_), 256, 0, stream>>>(Q, Kv, Vv, ctx);
  proj_gemm_kernel<<<dim3(128, 6), 256, 0, stream>>>(ctx, wprojb, proj_b, out);
}
